// MultiHeadAttention_76527727280338
// MI455X (gfx1250) — matrix operand order repaired and checked by reading, not compiled, not run on hardware
//
#include <hip/hip_runtime.h>
#include <hip/hip_bf16.h>

typedef _Float16 h4   __attribute__((ext_vector_type(4)));
typedef _Float16 h8   __attribute__((ext_vector_type(8)));
typedef _Float16 v16h __attribute__((ext_vector_type(16)));
typedef float    v8f  __attribute__((ext_vector_type(8)));
typedef float    f4   __attribute__((ext_vector_type(4)));
typedef int      b128v __attribute__((vector_size(16)));   // int4 payload type

union HFrag { v16h v; h8 h[2]; };

__device__ __forceinline__ v8f wmma_f16(const HFrag& a, const HFrag& b, v8f c) {
  // D = A(16x32 f16) * B(32x16 f16) + C(16x16 f32)
  return __builtin_amdgcn_wmma_f32_16x16x32_f16(
      /*neg_a=*/false, a.v, /*neg_b=*/false, b.v,
      /*c_mod=*/(short)0, c, /*reuse_a=*/false, /*reuse_b=*/false);
}

// ---- DPP16 cross-lane helpers (all reductions stay inside a 16-lane row) ----
// 0xB1 = quad_perm(1,0,3,2) -> xor 1 ; 0x4E = quad_perm(2,3,0,1) -> xor 2
// 0x141 = row_half_mirror   -> xor 4 ; 0x140 = row_mirror        -> xor 8
template<int CTRL, int XORM>
__device__ __forceinline__ float dpp_f32(float x) {
#if defined(__gfx1250__) && __has_builtin(__builtin_amdgcn_update_dpp)
  int i = __builtin_amdgcn_update_dpp(0, __builtin_bit_cast(int, x),
                                      CTRL, 0xF, 0xF, true);
  return __builtin_bit_cast(float, i);
#else
  return __shfl_xor(x, XORM, 32);
#endif
}
__device__ __forceinline__ float row16_max(float x) {
  x = fmaxf(x, dpp_f32<0xB1, 1>(x));
  x = fmaxf(x, dpp_f32<0x4E, 2>(x));
  x = fmaxf(x, dpp_f32<0x141, 4>(x));
  x = fmaxf(x, dpp_f32<0x140, 8>(x));
  return x;
}
__device__ __forceinline__ float row16_sum(float x) {
  x += dpp_f32<0xB1, 1>(x);
  x += dpp_f32<0x4E, 2>(x);
  x += dpp_f32<0x141, 4>(x);
  x += dpp_f32<0x140, 8>(x);
  return x;
}

// ---- CDNA5 async global->LDS copy (ASYNCcnt path), guarded fallback ----
__device__ __forceinline__ void async_copy_b128(const _Float16* g, _Float16* l) {
#if defined(__gfx1250__) && __has_builtin(__builtin_amdgcn_global_load_async_to_lds_b128)
  __builtin_amdgcn_global_load_async_to_lds_b128(
      (__attribute__((address_space(1))) b128v*)(_Float16*)g,
      (__attribute__((address_space(3))) b128v*)l, 0, 0);
#else
  *(h8*)l = *(const h8*)g;
#endif
}
__device__ __forceinline__ void async_wait0() {
#if defined(__gfx1250__) && __has_builtin(__builtin_amdgcn_s_wait_asynccnt)
  __builtin_amdgcn_s_wait_asynccnt(0);
#elif defined(__gfx1250__)
  asm volatile("s_wait_asynccnt 0" ::: "memory");
#endif
}

// ---------------------------------------------------------------------------
// Tiled WMMA GEMM: C[M,N] = A[M,K] * B[K,N] + bias[N]
// A: f32 (global input) or f16 (workspace); B/bias f32; C f16 (ws) or f32 (out)
// Block tile 128x128x32, 256 threads = 8 waves, wave tile 32x64 (2x4 WMMA tiles)
// ---------------------------------------------------------------------------
template<bool A_IS_F32, bool OUT_F16>
__global__ __launch_bounds__(256, 2) void gemm_wmma_kernel(
    const void* __restrict__ Aptr,
    const float* __restrict__ Bw,
    const float* __restrict__ bias,
    void* __restrict__ Cptr,
    int M, int N, int K)
{
  constexpr int BM = 128, BN = 128, BK = 32, LS = 40; // LS: f16 row stride (pad)
  __shared__ __align__(16) _Float16 As[BM * LS];      // [m][k] f16
  __shared__ __align__(16) _Float16 Bs[BN * LS];      // [n][k] f16 (transposed)

  const int tid   = threadIdx.x;
  const int lane  = tid & 31;
  const int wave  = tid >> 5;          // 0..7
  const int l15   = lane & 15;
  const int lhalf = lane >> 4;
  const int wm    = wave >> 1;         // 0..3 -> 32-row slab
  const int wn    = wave & 1;          // 0..1 -> 64-col slab
  const int bm    = blockIdx.y * BM;
  const int bn    = blockIdx.x * BN;

  v8f acc[2][4];
  #pragma unroll
  for (int mt = 0; mt < 2; ++mt)
    #pragma unroll
    for (int nt = 0; nt < 4; ++nt)
      acc[mt][nt] = (v8f){0.f, 0.f, 0.f, 0.f, 0.f, 0.f, 0.f, 0.f};

  for (int k0 = 0; k0 < K; k0 += BK) {
    // ---- stage A tile (128x32) into LDS as f16 [m][k] ----
    if constexpr (A_IS_F32) {
      const float* A = (const float*)Aptr;
      #pragma unroll
      for (int i = 0; i < 4; ++i) {                 // 1024 float4 / 256 thr
        int idx = tid + i * 256;
        int r = idx >> 3, c4 = (idx & 7) * 4;
        f4 s = *(const f4*)(A + (size_t)(bm + r) * K + k0 + c4);
        h4 d; d[0] = (_Float16)s[0]; d[1] = (_Float16)s[1];
              d[2] = (_Float16)s[2]; d[3] = (_Float16)s[3];
        *(h4*)(&As[r * LS + c4]) = d;
      }
    } else {
      const _Float16* A = (const _Float16*)Aptr;
      #pragma unroll
      for (int i = 0; i < 2; ++i) {                 // 512 b128 / 256 thr (async)
        int idx = tid + i * 256;
        int r = idx >> 2, c8 = (idx & 3) * 8;
        async_copy_b128(A + (size_t)(bm + r) * K + k0 + c8, &As[r * LS + c8]);
      }
    }
    // ---- stage B tile (32x128) into LDS transposed as f16 [n][k] ----
    #pragma unroll
    for (int i = 0; i < 4; ++i) {                   // 1024 float4 / 256 thr
      int idx = tid + i * 256;
      int kr = idx >> 5, c4 = (idx & 31) * 4;
      f4 w = *(const f4*)(Bw + (size_t)(k0 + kr) * N + bn + c4);
      #pragma unroll
      for (int j = 0; j < 4; ++j)
        Bs[(c4 + j) * LS + kr] = (_Float16)w[j];
    }
    // ---- prefetch next K-tile into L2 while we compute this one ----
    if (k0 + BK < K) {
      if constexpr (A_IS_F32)
        __builtin_prefetch((const float*)Aptr + (size_t)(bm + (tid >> 1)) * K +
                           k0 + BK + (tid & 1) * 16, 0, 0);
      else
        __builtin_prefetch((const _Float16*)Aptr + (size_t)(bm + (tid >> 1)) * K +
                           k0 + BK + (tid & 1) * 16, 0, 0);
      __builtin_prefetch(Bw + (size_t)(k0 + BK + (tid >> 3)) * N + bn +
                         (tid & 7) * 16, 0, 0);
    }
    if constexpr (!A_IS_F32) async_wait0();
    __syncthreads();

    // ---- fragments ----
    HFrag af[2], bf[4];
    const int arow = wm * 32 + l15;
    const int koff = lhalf * 8;       // A: lanes16-31 start at K=8 within 16
    #pragma unroll
    for (int mt = 0; mt < 2; ++mt) {
      const _Float16* p = &As[(arow + mt * 16) * LS];
      af[mt].h[0] = *(const h8*)(p + koff);         // K in [koff, koff+8)
      af[mt].h[1] = *(const h8*)(p + 16 + koff);    // K in [16+koff, ...)
    }
    const int bcol = wn * 64 + l15;
    const int kb2  = lhalf * 8;       // B: lanes16-31 start at K=8 within 16
    #pragma unroll
    for (int nt = 0; nt < 4; ++nt) {
      const _Float16* p = &Bs[(bcol + nt * 16) * LS + kb2];
      bf[nt].h[0] = *(const h8*)(p);
      bf[nt].h[1] = *(const h8*)(p + 16);
    }
    #pragma unroll
    for (int mt = 0; mt < 2; ++mt)
      #pragma unroll
      for (int nt = 0; nt < 4; ++nt)
        acc[mt][nt] = wmma_f16(af[mt], bf[nt], acc[mt][nt]);
    __syncthreads();
  }

  // ---- epilogue: C-layout (row = vgpr + 8*lhalf, col = l15) ----
  #pragma unroll
  for (int nt = 0; nt < 4; ++nt) {
    const int col = bn + wn * 64 + nt * 16 + l15;
    const float bv = bias[col];
    #pragma unroll
    for (int mt = 0; mt < 2; ++mt) {
      #pragma unroll
      for (int v = 0; v < 8; ++v) {
        const int row = bm + wm * 32 + mt * 16 + lhalf * 8 + v;
        const float val = acc[mt][nt][v] + bv;
        if constexpr (OUT_F16)
          ((_Float16*)Cptr)[(size_t)row * N + col] = (_Float16)val;
        else
          ((float*)Cptr)[(size_t)row * N + col] = val;
      }
    }
  }
}

// ---------------------------------------------------------------------------
// Flash attention (causal): one block per (b*H+h, qblock of 64 rows).
// 128 threads = 4 waves; wave w owns q rows [q0+16w, q0+16w+16).
// qkv: f16 [B*S, 3*768]; ao: f16 [B*S, 768]
// ---------------------------------------------------------------------------
__global__ __launch_bounds__(128, 4) void attn_flash_kernel(
    const _Float16* __restrict__ qkv, _Float16* __restrict__ ao)
{
  constexpr int S = 2048, E3 = 2304, E = 768, D = 64, NH = 12;
  const int bh   = blockIdx.x;           // 0..47
  const int b    = bh / NH, h = bh % NH;
  const int q0   = blockIdx.y * 64;
  const int tid  = threadIdx.x;
  const int lane = tid & 31, wave = tid >> 5;
  const int l15  = lane & 15, lhalf = lane >> 4;

  __shared__ __align__(16) _Float16 Qs[64 * 72];     // [q][d]
  __shared__ __align__(16) _Float16 Ks[32 * 72];     // [key][d]
  __shared__ __align__(16) _Float16 Vt[64 * 40];     // [d][key] (transposed)
  __shared__ __align__(16) _Float16 Ps[4 * 16 * 40]; // per-wave P scratch [q][key]

  const _Float16* qb  = qkv + (size_t)(b * S) * E3 + h * D;       // Q cols
  const _Float16* kb_ = qb + E;                                   // K cols
  const _Float16* vb  = qb + 2 * E;                               // V cols

  // stage Q (64x64 halfs = 512 b128; 128 thr -> 4 each) via async path
  #pragma unroll
  for (int i = 0; i < 4; ++i) {
    int idx = tid + i * 128;
    int r = idx >> 3, c8 = (idx & 7) * 8;
    async_copy_b128(qb + (size_t)(q0 + r) * E3 + c8, &Qs[r * 72 + c8]);
  }
  async_wait0();
  __syncthreads();

  // resident Q A-frags: aq[0] -> d 0..31, aq[1] -> d 32..63
  HFrag aq[2];
  {
    const _Float16* p = &Qs[(wave * 16 + l15) * 72];
    const int koff = lhalf * 8;
    aq[0].h[0] = *(const h8*)(p + koff);       aq[0].h[1] = *(const h8*)(p + 16 + koff);
    aq[1].h[0] = *(const h8*)(p + 32 + koff);  aq[1].h[1] = *(const h8*)(p + 48 + koff);
  }

  float mrow[8], lrow[8];
  v8f accd[4];
  #pragma unroll
  for (int v = 0; v < 8; ++v) { mrow[v] = -1e30f; lrow[v] = 0.f; }
  #pragma unroll
  for (int dt = 0; dt < 4; ++dt)
    accd[dt] = (v8f){0.f, 0.f, 0.f, 0.f, 0.f, 0.f, 0.f, 0.f};

  const int kend = q0 + 64;                      // causal bound for this block
  for (int kblk = 0; kblk < kend; kblk += 32) {
    __syncthreads();                             // protect prev-iter frag reads
    // stage K tile [32 keys][64 d] (256 b128; 2 each) via async path
    #pragma unroll
    for (int i = 0; i < 2; ++i) {
      int idx = tid + i * 128;
      int r = idx >> 3, c8 = (idx & 7) * 8;
      async_copy_b128(kb_ + (size_t)(kblk + r) * E3 + c8, &Ks[r * 72 + c8]);
    }
    // stage V transposed -> Vt[d][key] (512 h4; 4 each)
    #pragma unroll
    for (int i = 0; i < 4; ++i) {
      int idx = tid + i * 128;
      int r = idx >> 4, c4 = (idx & 15) * 4;
      h4 vv = *(const h4*)(vb + (size_t)(kblk + r) * E3 + c4);
      #pragma unroll
      for (int j = 0; j < 4; ++j) Vt[(c4 + j) * 40 + r] = vv[j];
    }
    async_wait0();
    __syncthreads();

    // ---- scores: two 16-key subtiles; scale + causal mask ----
    v8f pt[2];
    #pragma unroll
    for (int nt = 0; nt < 2; ++nt) {
      HFrag bk0, bk1;
      const _Float16* p = &Ks[(nt * 16 + l15) * 72];
      const int d2 = lhalf * 8;
      bk0.h[0] = *(const h8*)(p + d2);       bk0.h[1] = *(const h8*)(p + d2 + 16);
      bk1.h[0] = *(const h8*)(p + 32 + d2);  bk1.h[1] = *(const h8*)(p + 32 + d2 + 16);
      v8f s = (v8f){0.f, 0.f, 0.f, 0.f, 0.f, 0.f, 0.f, 0.f};
      s = wmma_f16(aq[0], bk0, s);
      s = wmma_f16(aq[1], bk1, s);
      const int kg = kblk + nt * 16 + l15;
      #pragma unroll
      for (int v = 0; v < 8; ++v) {
        const int qg = q0 + wave * 16 + lhalf * 8 + v;
        float val = s[v] * 0.125f;             // 1/sqrt(64)
        if (kg > qg) val = -1e30f;
        s[v] = val;
      }
      pt[nt] = s;
    }

    // ---- online softmax (DPP row reductions, no LDS traffic) ----
    float alpha[8];
    #pragma unroll
    for (int v = 0; v < 8; ++v) {
      const float mx = row16_max(fmaxf(pt[0][v], pt[1][v]));
      const float mnew = fmaxf(mrow[v], mx);
      alpha[v] = __expf(mrow[v] - mnew);
      mrow[v] = mnew;
    }
    #pragma unroll
    for (int nt = 0; nt < 2; ++nt)
      #pragma unroll
      for (int v = 0; v < 8; ++v)
        pt[nt][v] = __expf(pt[nt][v] - mrow[v]);
    #pragma unroll
    for (int v = 0; v < 8; ++v) {
      const float sum = row16_sum(pt[0][v] + pt[1][v]);
      lrow[v] = lrow[v] * alpha[v] + sum;
    }
    #pragma unroll
    for (int dt = 0; dt < 4; ++dt)
      #pragma unroll
      for (int v = 0; v < 8; ++v)
        accd[dt][v] *= alpha[v];

    // ---- C-layout -> A-layout remap of P via per-wave LDS ----
    _Float16* pw = &Ps[wave * 16 * 40];
    #pragma unroll
    for (int nt = 0; nt < 2; ++nt)
      #pragma unroll
      for (int v = 0; v < 8; ++v)
        pw[(lhalf * 8 + v) * 40 + nt * 16 + l15] = (_Float16)pt[nt][v];
    asm volatile("s_wait_dscnt 0" ::: "memory");   // cross-lane LDS RAW in-wave

    HFrag ap;
    {
      const _Float16* p = &Ps[wave * 16 * 40 + l15 * 40];
      const int koff = lhalf * 8;
      ap.h[0] = *(const h8*)(p + koff);
      ap.h[1] = *(const h8*)(p + 16 + koff);
    }
    // ---- accd += P * V ----
    #pragma unroll
    for (int dt = 0; dt < 4; ++dt) {
      HFrag bv;
      const _Float16* p = &Vt[(dt * 16 + l15) * 40 + lhalf * 8];
      bv.h[0] = *(const h8*)(p);
      bv.h[1] = *(const h8*)(p + 16);
      accd[dt] = wmma_f16(ap, bv, accd[dt]);
    }
  }

  // ---- epilogue: out = acc / l, f16 into [B*S, 768] at head column ----
  #pragma unroll
  for (int dt = 0; dt < 4; ++dt) {
    #pragma unroll
    for (int v = 0; v < 8; ++v) {
      const int row  = q0 + wave * 16 + lhalf * 8 + v;
      const int dcol = dt * 16 + l15;
      const float val = accd[dt][v] / lrow[v];
      ao[(size_t)(b * S + row) * E + h * D + dcol] = (_Float16)val;
    }
  }
}

// ---------------------------------------------------------------------------
extern "C" void kernel_launch(void* const* d_in, const int* in_sizes, int n_in,
                              void* d_out, int out_size, void* d_ws, size_t ws_size,
                              hipStream_t stream) {
  (void)in_sizes; (void)n_in; (void)out_size; (void)ws_size;
  const float* x      = (const float*)d_in[0];   // [4,2048,768]
  const float* w_attn = (const float*)d_in[1];   // [768,2304]
  const float* b_attn = (const float*)d_in[2];   // [2304]
  const float* w_proj = (const float*)d_in[3];   // [768,768]
  const float* b_proj = (const float*)d_in[4];   // [768]
  float* out = (float*)d_out;                    // [4,2048,768] f32

  constexpr int M = 4 * 2048;                    // 8192 rows
  _Float16* qkv_h = (_Float16*)d_ws;                                 // 8192x2304 f16
  _Float16* ao_h  = (_Float16*)((char*)d_ws + (size_t)M * 2304 * 2); // 8192x768 f16

  // 1) QKV projection: f32 in -> f16 qkv
  dim3 g1(2304 / 128, M / 128);
  gemm_wmma_kernel<true, true><<<g1, 256, 0, stream>>>(
      x, w_attn, b_attn, qkv_h, M, 2304, 768);

  // 2) causal flash attention per (b,h, 64-query block)
  dim3 g2(4 * 12, 2048 / 64);
  attn_flash_kernel<<<g2, 128, 0, stream>>>(qkv_h, ao_h);

  // 3) output projection: f16 in -> f32 out (+bias)
  dim3 g3(768 / 128, M / 128);
  gemm_wmma_kernel<false, false><<<g3, 256, 0, stream>>>(
      ao_h, w_proj, b_proj, out, M, 768, 768);
}
